// SpatialSelfAttention_75883482186325
// MI455X (gfx1250) — matrix operand order repaired and checked by reading, not compiled, not run on hardware
//
#include <hip/hip_runtime.h>
#include <hip/hip_bf16.h>

typedef __bf16 bf16;
typedef __attribute__((ext_vector_type(16))) __bf16 v16bf;
typedef __attribute__((ext_vector_type(8)))  float  v8f;
typedef __attribute__((ext_vector_type(4)))  unsigned int u32x4;

#define B_    8
#define C_    256
#define N_    2304
#define CQK_  64

union FragU { u32x4 q[2]; v16bf v; };
union Pack8 { bf16 h[8]; u32x4 u; };

// A fragment: 16x32 bf16, A row-major with lda = K.
// Lane l holds row (l&15); halves 0-7 = K kk+(l<16?0:8)..+7, halves 8-15 = +16.
__device__ __forceinline__ v16bf load_a_frag(const bf16* __restrict__ A, int lda,
                                             int row, int kk, int lane) {
  const int r  = row + (lane & 15);
  const int k0 = kk + ((lane < 16) ? 0 : 8);
  const bf16* p = A + (long long)r * lda + k0;
  FragU u;
  u.q[0] = *(const u32x4*)(p);        // halves 0-7
  u.q[1] = *(const u32x4*)(p + 16);   // halves 8-15 (K +16)
  return u.v;
}

// B fragment: 32x16 bf16, B stored column-major (per output column, K contiguous), ldb = K.
// Lane l holds column (l&15); halves 0-7 = K kk+(l<16?0:8)..+7, halves 8-15 = +16.
__device__ __forceinline__ v16bf load_b_frag(const bf16* __restrict__ Bc, int ldb,
                                             int col, int kk, int lane) {
  const int c  = col + (lane & 15);
  const int k0 = kk + ((lane < 16) ? 0 : 8);
  const bf16* p = Bc + (long long)c * ldb + k0;
  FragU u;
  u.q[0] = *(const u32x4*)(p);
  u.q[1] = *(const u32x4*)(p + 16);
  return u.v;
}

// MODE 0: D = f32 row-major [m][n]           (optional bias[m])
// MODE 1: D = bf16 row-major [m][n]          (optional bias[m])
// MODE 2: D = bf16 transposed [n][m] packed  (optional bias[m])
// A: bf16 row-major [M][K]; Bc: bf16 column-major [N][K].
// Wave tile = 32(M) x 64(N): 8 f32 accumulators, B fragments reused across both M rows.
// Consecutive waves in a block share n0 (when mTiles >= 8, a block covers a full
// 256-row M strip with identical B addresses -> WGP$ hits for 7/8 of B traffic).
template<int MODE>
__global__ __launch_bounds__(256) void gemm_kernel(
    const bf16* __restrict__ A, const bf16* __restrict__ Bc,
    const float* __restrict__ bias, void* __restrict__ Dv,
    int M, int N, int K, int ldd,
    long long strideA, long long strideB, long long strideD)
{
  const int batch = blockIdx.y;
  const int lane  = threadIdx.x & 31;
  const int w     = threadIdx.x >> 5;
  const int mTiles  = M >> 5;           // 32-row wave tiles
  const int nStrips = N >> 6;
  const int waveTile = blockIdx.x * 8 + w;
  if (waveTile >= mTiles * nStrips) return;   // uniform per wave: EXEC stays all-ones
  const int m0 = (waveTile % mTiles) << 5;
  const int n0 = (waveTile / mTiles) << 6;

  const bf16* Ab = A  + (long long)batch * strideA;
  const bf16* Bb = Bc + (long long)batch * strideB;

  v8f acc[2][4];
  #pragma unroll
  for (int mi = 0; mi < 2; ++mi)
    #pragma unroll
    for (int j = 0; j < 4; ++j)
      acc[mi][j] = (v8f){0.f,0.f,0.f,0.f,0.f,0.f,0.f,0.f};

  for (int kk = 0; kk < K; kk += 32) {
    v16bf a0 = load_a_frag(Ab, K, m0,      kk, lane);
    v16bf a1 = load_a_frag(Ab, K, m0 + 16, kk, lane);
    v16bf b0 = load_b_frag(Bb, K, n0 +  0, kk, lane);
    v16bf b1 = load_b_frag(Bb, K, n0 + 16, kk, lane);
    v16bf b2 = load_b_frag(Bb, K, n0 + 32, kk, lane);
    v16bf b3 = load_b_frag(Bb, K, n0 + 48, kk, lane);
    acc[0][0] = __builtin_amdgcn_wmma_f32_16x16x32_bf16(false, a0, false, b0, (short)0, acc[0][0], false, false);
    acc[1][0] = __builtin_amdgcn_wmma_f32_16x16x32_bf16(false, a1, false, b0, (short)0, acc[1][0], false, false);
    acc[0][1] = __builtin_amdgcn_wmma_f32_16x16x32_bf16(false, a0, false, b1, (short)0, acc[0][1], false, false);
    acc[1][1] = __builtin_amdgcn_wmma_f32_16x16x32_bf16(false, a1, false, b1, (short)0, acc[1][1], false, false);
    acc[0][2] = __builtin_amdgcn_wmma_f32_16x16x32_bf16(false, a0, false, b2, (short)0, acc[0][2], false, false);
    acc[1][2] = __builtin_amdgcn_wmma_f32_16x16x32_bf16(false, a1, false, b2, (short)0, acc[1][2], false, false);
    acc[0][3] = __builtin_amdgcn_wmma_f32_16x16x32_bf16(false, a0, false, b3, (short)0, acc[0][3], false, false);
    acc[1][3] = __builtin_amdgcn_wmma_f32_16x16x32_bf16(false, a1, false, b3, (short)0, acc[1][3], false, false);
  }

  const int colInTile = lane & 15;
  const int rowHalf   = (lane < 16) ? 0 : 8;   // D: VGPR i -> row base+i, col lane&15

  #pragma unroll
  for (int mi = 0; mi < 2; ++mi) {
    const int rowBase = m0 + mi * 16 + rowHalf;
    #pragma unroll
    for (int j = 0; j < 4; ++j) {
      const int n = n0 + j * 16 + colInTile;
      if (MODE == 0) {
        float* D = (float*)Dv + (long long)batch * strideD;
        #pragma unroll
        for (int i = 0; i < 8; ++i) {
          float vv = acc[mi][j][i];
          if (bias) vv += bias[rowBase + i];
          D[(long long)(rowBase + i) * ldd + n] = vv;
        }
      } else if (MODE == 1) {
        bf16* D = (bf16*)Dv + (long long)batch * strideD;
        #pragma unroll
        for (int i = 0; i < 8; ++i) {
          float vv = acc[mi][j][i];
          if (bias) vv += bias[rowBase + i];
          D[(long long)(rowBase + i) * ldd + n] = (bf16)vv;
        }
      } else {
        bf16* D = (bf16*)Dv + (long long)batch * strideD;
        Pack8 pk;
        #pragma unroll
        for (int i = 0; i < 8; ++i) {
          float vv = acc[mi][j][i];
          if (bias) vv += bias[rowBase + i];
          pk.h[i] = (bf16)vv;
        }
        *(u32x4*)(D + (long long)n * ldd + rowBase) = pk.u;   // 8 contiguous bf16
      }
    }
  }
}

// features f32 [B][C][N] -> xT bf16 [B][N][C] via LDS-tiled transpose (coalesced both sides)
__global__ __launch_bounds__(256) void transpose_x_kernel(const float* __restrict__ src,
                                                          bf16* __restrict__ dst) {
  __shared__ float tile[32][33];
  const int b  = blockIdx.z;
  const int n0 = blockIdx.x * 32;
  const int c0 = blockIdx.y * 32;
  const int tx = threadIdx.x;       // 0..31
  const int ty = threadIdx.y;       // 0..7
  #pragma unroll
  for (int i = 0; i < 4; ++i) {
    const int c = c0 + ty + i * 8;
    tile[ty + i * 8][tx] = src[((long long)b * C_ + c) * N_ + n0 + tx];
  }
  __syncthreads();
  #pragma unroll
  for (int i = 0; i < 4; ++i) {
    const int n = n0 + ty + i * 8;
    dst[((long long)b * N_ + n) * C_ + c0 + tx] = (bf16)tile[tx][ty + i * 8];
  }
}

__global__ __launch_bounds__(256) void f32_to_bf16_kernel(const float* __restrict__ src,
                                                          bf16* __restrict__ dst, int n) {
  const int i = blockIdx.x * blockDim.x + threadIdx.x;
  if (i < n) dst[i] = (bf16)src[i];
}

// Softmax over the batch axis, in place on bf16 logits [B][N*N] (coalesced, elementwise)
__global__ __launch_bounds__(256) void softmax_batch_kernel(bf16* __restrict__ logits) {
  const long long sites = (long long)N_ * N_;
  const long long j = (long long)blockIdx.x * blockDim.x + threadIdx.x;
  if (j >= sites) return;
  float l[B_];
  float mx = -3.0e38f;
  #pragma unroll
  for (int b = 0; b < B_; ++b) {
    l[b] = (float)logits[(long long)b * sites + j];
    mx = fmaxf(mx, l[b]);
  }
  float s = 0.f;
  #pragma unroll
  for (int b = 0; b < B_; ++b) { l[b] = __expf(l[b] - mx); s += l[b]; }
  const float inv = 1.f / s;
  #pragma unroll
  for (int b = 0; b < B_; ++b)
    logits[(long long)b * sites + j] = (bf16)(l[b] * inv);
}

extern "C" void kernel_launch(void* const* d_in, const int* in_sizes, int n_in,
                              void* d_out, int out_size, void* d_ws, size_t ws_size,
                              hipStream_t stream) {
  const float* features = (const float*)d_in[0];
  const float* Wq = (const float*)d_in[1];
  const float* bq = (const float*)d_in[2];
  const float* Wk = (const float*)d_in[3];
  const float* bk = (const float*)d_in[4];
  const float* Wv = (const float*)d_in[5];
  const float* bv = (const float*)d_in[6];

  char* ws = (char*)d_ws;
  // workspace layout (bytes), all offsets 256B aligned
  bf16* wq_b = (bf16*)(ws + 0);                       //  64*256*2 =     32768
  bf16* wk_b = (bf16*)(ws + 32768);                   //                 32768
  bf16* wv_b = (bf16*)(ws + 65536);                   // 256*256*2 =    131072
  bf16* xT   = (bf16*)(ws + 196608);                  // B*N*C*2   =   9437184
  bf16* qT   = (bf16*)(ws + 9633792);                 // B*N*64*2  =   2359296
  bf16* kT   = (bf16*)(ws + 11993088);                // B*N*64*2  =   2359296
  bf16* vB   = (bf16*)(ws + 14352384);                // B*256*N*2 =   9437184
  bf16* lg   = (bf16*)(ws + 23789568);                // B*N*N*2   =  84934656  (end 108724224)

  // 1) weights -> bf16 (row-major, K contiguous == A-fragment layout)
  f32_to_bf16_kernel<<<64,  256, 0, stream>>>(Wq, wq_b, 64 * 256);
  f32_to_bf16_kernel<<<64,  256, 0, stream>>>(Wk, wk_b, 64 * 256);
  f32_to_bf16_kernel<<<256, 256, 0, stream>>>(Wv, wv_b, 256 * 256);

  // 2) x -> xT bf16 [b][n][c]  (column-major B operand for the projections)
  transpose_x_kernel<<<dim3(N_ / 32, C_ / 32, B_), dim3(32, 8), 0, stream>>>(features, xT);

  // 3) projections (A = W shared across batch, strideA = 0)
  //    q,k stored transposed [n][64] -> directly usable as B (q) / A (k) of the logits GEMM
  //    wave tiles are 32(M) x 64(N): tiles = (M/32)*(N/64), blocks = tiles/8
  gemm_kernel<2><<<dim3(( 2 * 36) / 8, B_), 256, 0, stream>>>(
      wq_b, xT, bq, qT, 64, N_, 256, CQK_, 0LL, (long long)N_ * C_, (long long)N_ * CQK_);
  gemm_kernel<2><<<dim3(( 2 * 36) / 8, B_), 256, 0, stream>>>(
      wk_b, xT, bk, kT, 64, N_, 256, CQK_, 0LL, (long long)N_ * C_, (long long)N_ * CQK_);
  gemm_kernel<1><<<dim3(( 8 * 36) / 8, B_), 256, 0, stream>>>(
      wv_b, xT, bv, vB, 256, N_, 256, N_, 0LL, (long long)N_ * C_, (long long)C_ * N_);

  // 4) logits: S^T[b][m][n] = sum_c k[b][m][c] * q[b][n][c]   (K = 64)
  gemm_kernel<1><<<dim3((72 * 36) / 8, B_), 256, 0, stream>>>(
      kT, qT, nullptr, lg, N_, N_, CQK_, N_,
      (long long)N_ * CQK_, (long long)N_ * CQK_, (long long)N_ * N_);

  // 5) softmax over batch dim, in place (each thread owns one (m,n) site across all 8 batches)
  softmax_batch_kernel<<<(N_ * N_) / 256, 256, 0, stream>>>(lg);

  // 6) out[b][c][m] = sum_n v[b][c][n] * attn^T[b][m][n]  -> d_out f32 [B][C][H][W]
  //    mTiles = 8 -> each block's 8 waves cover the full M strip sharing B fragments
  gemm_kernel<0><<<dim3(( 8 * 36) / 8, B_), 256, 0, stream>>>(
      vB, lg, nullptr, d_out, C_, N_, N_, N_,
      (long long)C_ * N_, (long long)N_ * N_, (long long)C_ * N_);
}
